// GCNLayer_15358803050969
// MI455X (gfx1250) — compile-verified, matrix operand order checked by reading, not run on hardware
//
#include <hip/hip_runtime.h>

#define NPTS 8192
#define BATCH 4
#define CIN 64
#define COUT 64
#define KNN 8
#define TILE 1024
#define NTILE (NPTS / TILE)

typedef __attribute__((ext_vector_type(16))) __bf16        v16bf;
typedef __attribute__((ext_vector_type(8)))  float         v8f;
typedef __attribute__((ext_vector_type(8)))  unsigned int  v8u;
typedef __attribute__((ext_vector_type(4)))  unsigned int  v4u;
typedef __attribute__((ext_vector_type(8)))  int           v8i;
typedef __attribute__((ext_vector_type(4)))  int           v4i;

__device__ __forceinline__ unsigned short f2bf(float f) {
  unsigned int u = __builtin_bit_cast(unsigned int, f);
  u += 0x7FFFu + ((u >> 16) & 1u);      // round-to-nearest-even
  return (unsigned short)(u >> 16);
}
__device__ __forceinline__ unsigned int pack_bf(float lo, float hi) {
  return ((unsigned int)f2bf(hi) << 16) | (unsigned int)f2bf(lo);
}

// TDM: DMA one [3][TILE] fp32 tile (rows x/y/z of xyz, row stride NPTS) into LDS.
// Descriptor per CDNA5 ISA ch.8: group0 {count=1, lds_addr, global_addr, type=2},
// group1 {data_size=4B, tensor_dim0=NPTS, tensor_dim1=3, tile_dim0=TILE,
// tile_dim1=3, tensor_dim0_stride=NPTS}; groups 2/3 zero (2-D tensor, no cluster).
__device__ __forceinline__ void tdm_load_xyz_tile(unsigned lds_off,
                                                  unsigned long long gaddr) {
  v4u g0;
  g0[0] = 1u;                                    // count=1, user descriptor
  g0[1] = lds_off;                               // lds_addr (bytes)
  g0[2] = (unsigned)gaddr;                       // global_addr[31:0]
  g0[3] = (unsigned)((gaddr >> 32) & 0x01FFFFFFull) | 0x80000000u; // [56:32] | type=2
  v8i g1;
  g1[0] = 0x00020000;                            // wg_mask=0, data_size=2 (4B)
  g1[1] = (int)((unsigned)NPTS << 16);           // tensor_dim0[15:0] in [31:16]
  g1[2] = (int)(3u << 16);                       // tensor_dim1[15:0]=3 in [31:16]
  g1[3] = (int)((unsigned)TILE << 16);           // tile_dim0 in [31:16]
  g1[4] = 3;                                     // tile_dim1=3, tile_dim2=0
  g1[5] = NPTS;                                  // tensor_dim0_stride[31:0]
  g1[6] = 0;
  g1[7] = 0;
  v4i z4 = {0, 0, 0, 0};
#if __clang_major__ >= 23
  v8i z8 = {0, 0, 0, 0, 0, 0, 0, 0};
  __builtin_amdgcn_tensor_load_to_lds(g0, g1, z4, z4, z8, 0);
#else
  __builtin_amdgcn_tensor_load_to_lds(g0, g1, z4, z4, 0);
#endif
}

// ---------------- Kernel 1: KNN (top-8 of -dist^2, self included) ----------
// TDM double-buffers xyz tiles into LDS while all 8 waves compute distances.
__global__ __launch_bounds__(256) void knn_kernel(const float* __restrict__ xyz,
                                                  int* __restrict__ idx_out) {
  __shared__ float sbuf[2][3][TILE];             // 24 KB double buffer
  int g = blockIdx.x * 256 + threadIdx.x;        // b*N + n (blocks never straddle batches)
  int b = g >> 13;
  int n = g & (NPTS - 1);
  const float* X = xyz + (size_t)b * 3 * NPTS;
  float qx = X[n], qy = X[NPTS + n], qz = X[2 * NPTS + n];

  float best[KNN]; int bid[KNN];
#pragma unroll
  for (int j = 0; j < KNN; ++j) { best[j] = -3.4e38f; bid[j] = 0; }

  // Uniform (scalar-branch) issuer gate: TDM ignores EXEC, so only wave 0 may
  // reach the issue instruction at all.
  bool issuer = ((__builtin_amdgcn_readfirstlane((int)threadIdx.x) >> 5) == 0);
  unsigned lds0 = (unsigned)(size_t)(&sbuf[0][0][0]);
  unsigned lds1 = (unsigned)(size_t)(&sbuf[1][0][0]);

  if (issuer) tdm_load_xyz_tile(lds0, (unsigned long long)(size_t)X);

  for (int t = 0; t < NTILE; ++t) {
    if (issuer) {
      if (t + 1 < NTILE) {
        // prefetch next tile into the other buffer, then wait for current one
        tdm_load_xyz_tile((t & 1) ? lds0 : lds1,
                          (unsigned long long)(size_t)(X + (t + 1) * TILE));
        __builtin_amdgcn_s_wait_tensorcnt(1);    // in-order: tile t is resident
      } else {
        __builtin_amdgcn_s_wait_tensorcnt(0);
      }
    }
    __syncthreads();                             // publish buffer t to all waves
    const float* sx = sbuf[t & 1][0];
    const float* sy = sbuf[t & 1][1];
    const float* sz = sbuf[t & 1][2];
    int mbase = t * TILE;
    for (int mm = 0; mm < TILE; ++mm) {
      float dx = qx - sx[mm], dy = qy - sy[mm], dz = qz - sz[mm];
      float v = -(dx * dx + dy * dy + dz * dz);
      if (v > best[KNN - 1]) {                   // strict > keeps tie order = lower index
        best[KNN - 1] = v; bid[KNN - 1] = mbase + mm;
#pragma unroll
        for (int j = KNN - 1; j > 0; --j) {
          if (best[j] > best[j - 1]) {
            float tv = best[j]; best[j] = best[j - 1]; best[j - 1] = tv;
            int   ti = bid[j];  bid[j]  = bid[j - 1];  bid[j - 1]  = ti;
          }
        }
      }
    }
    __syncthreads();                             // buffer may be overwritten next iter
  }
#pragma unroll
  for (int j = 0; j < KNN; ++j)
    idx_out[(size_t)g * KNN + j] = bid[j];
}

// ------- Kernel 2: EdgeConv GEMM (bf16 WMMA) + fused BN/LeakyReLU/maxK -----
__global__ __launch_bounds__(256) void edgeconv_kernel(
    const float* __restrict__ inp, const int* __restrict__ knn_idx,
    const float* __restrict__ w,   const float* __restrict__ bn_g,
    const float* __restrict__ bn_b, const float* __restrict__ bn_m,
    const float* __restrict__ bn_v, float* __restrict__ out) {
  // Weights pre-packed into exact WMMA B-fragment layout: [ct][kb][j][lane]
  __shared__ unsigned int wlds[4 * 4 * 8 * 32];
  for (int e = threadIdx.x; e < 4096; e += 256) {
    int lane = e & 31, j = (e >> 5) & 7, kb = (e >> 8) & 3, ct = (e >> 10) & 3;
    int nc = lane & 15, hi = lane >> 4;
    int kk = ((j < 4) ? 0 : 16) + hi * 8 + (j & 3) * 2;   // B layout: K pair per VGPR
    int o = ct * 16 + nc, c = kb * 32 + kk;
    wlds[e] = pack_bf(w[o * 128 + c], w[o * 128 + c + 1]);
  }
  __syncthreads();

  int wid  = threadIdx.x >> 5;
  int lane = threadIdx.x & 31;
  int tile = blockIdx.x * 8 + wid;          // 16384 tiles of 16 rows
  int pair = tile * 2;                      // two points per tile
  int b  = pair >> 13;
  int n0 = pair & (NPTS - 1);
  int l0 = lane & 15, hi = lane >> 4;
  int noff = l0 >> 3, kn = l0 & 7;          // A row M = noff*8 + kn = l0
  int n = n0 + noff;
  int nbi = knn_idx[((size_t)b * NPTS + n) * KNN + kn];
  const float* cen = inp + ((size_t)b * NPTS + n)   * CIN;
  const float* nbr = inp + ((size_t)b * NPTS + nbi) * CIN;

  // A fragments for all 4 k-steps. Channels 0..63: nbr-center (kb 0,1);
  // channels 64..127: center (kb 2,3) — reuses the same center float2 loads.
  v8u a0 = {}, a1 = {}, a2 = {}, a3 = {};
#pragma unroll
  for (int kb = 0; kb < 2; ++kb) {
#pragma unroll
    for (int j = 0; j < 8; ++j) {
      int ch = kb * 32 + ((j < 4) ? 0 : 16) + hi * 8 + (j & 3) * 2;
      float2 cv = *(const float2*)(cen + ch);
      float2 nv = *(const float2*)(nbr + ch);
      unsigned int de = pack_bf(nv.x - cv.x, nv.y - cv.y);
      unsigned int ce = pack_bf(cv.x, cv.y);
      if (kb == 0) { a0[j] = de; a2[j] = ce; }
      else         { a1[j] = de; a3[j] = ce; }
    }
  }
  v16bf av[4];
  av[0] = __builtin_bit_cast(v16bf, a0);
  av[1] = __builtin_bit_cast(v16bf, a1);
  av[2] = __builtin_bit_cast(v16bf, a2);
  av[3] = __builtin_bit_cast(v16bf, a3);

#pragma unroll
  for (int ct = 0; ct < 4; ++ct) {
    v8f acc = {0.f, 0.f, 0.f, 0.f, 0.f, 0.f, 0.f, 0.f};
#pragma unroll
    for (int kb = 0; kb < 4; ++kb) {
      v8u bd;
#pragma unroll
      for (int j = 0; j < 8; ++j)
        bd[j] = wlds[((ct * 4 + kb) * 8 + j) * 32 + lane];  // bank-conflict-free
      acc = __builtin_amdgcn_wmma_f32_16x16x32_bf16(
          false, av[kb], false, __builtin_bit_cast(v16bf, bd),
          (short)0, acc, false, false);
    }
    // max over k: acc VGPR v holds M=v (lanes 0-15) / M=v+8 (lanes 16-31),
    // i.e. the 8 neighbors of point n0+hi at channel ct*16+l0.
    float mx = acc[0];
#pragma unroll
    for (int q = 1; q < 8; ++q) mx = fmaxf(mx, acc[q]);
    int o = ct * 16 + l0;
    float inv  = bn_g[o] * rsqrtf(bn_v[o] + 1e-3f);   // inv > 0 => BN/lrelu commute with max
    float bias = bn_b[o] - bn_m[o] * inv;
    float yv = mx * inv + bias;
    yv = (yv >= 0.f) ? yv : 0.2f * yv;
    out[((size_t)b * NPTS + (n0 + hi)) * COUT + o] = yv;
  }
}

extern "C" void kernel_launch(void* const* d_in, const int* in_sizes, int n_in,
                              void* d_out, int out_size, void* d_ws, size_t ws_size,
                              hipStream_t stream) {
  const float* inp = (const float*)d_in[0];
  const float* xyz = (const float*)d_in[1];
  const float* w   = (const float*)d_in[2];
  const float* g   = (const float*)d_in[3];
  const float* bb  = (const float*)d_in[4];
  const float* mn  = (const float*)d_in[5];
  const float* vr  = (const float*)d_in[6];
  float* out = (float*)d_out;
  int* idx = (int*)d_ws;                       // B*N*K ints = 1 MB scratch

  knn_kernel<<<(BATCH * NPTS) / 256, 256, 0, stream>>>(xyz, idx);
  edgeconv_kernel<<<(BATCH * NPTS * KNN) / (16 * 8), 256, 0, stream>>>(
      inp, idx, w, g, bb, mn, vr, out);
}
